// GCN_22170621182209
// MI455X (gfx1250) — matrix operand order repaired and checked by reading, not compiled, not run on hardware
//
#include <hip/hip_runtime.h>
#include <hip/hip_bf16.h>

#define NN 10000      // nodes
#define NE 320000     // edges
#define FD 256        // D == H == 256
#define APAD 8        // LDS row padding (bf16 elems) -> 528B pitch
#define APITCH (FD + APAD)

typedef __attribute__((ext_vector_type(16))) __bf16 v16bf;
typedef __attribute__((ext_vector_type(8)))  float  v8f;

// ---------------- utility kernels ----------------

__global__ void k_zero(float* __restrict__ p, int n) {
    int i = blockIdx.x * blockDim.x + threadIdx.x;
    if (i < n) p[i] = 0.0f;
}

__global__ void k_cvt_bf16(const float* __restrict__ in, __bf16* __restrict__ out, int n) {
    int i = blockIdx.x * blockDim.x + threadIdx.x;
    if (i < n) out[i] = (__bf16)in[i];
}

// deg[dst] += 1 for every edge (self-loop handled as +1 later)
__global__ void k_degree(const int* __restrict__ dst, float* __restrict__ deg, int e) {
    int i = blockIdx.x * blockDim.x + threadIdx.x;
    if (i < e) unsafeAtomicAdd(deg + dst[i], 1.0f);
}

// dinv[i] = rsqrt(deg[i] + 1)   (the +1 is the self-loop)
__global__ void k_dinv(float* __restrict__ deg_dinv, int n) {
    int i = blockIdx.x * blockDim.x + threadIdx.x;
    if (i < n) deg_dinv[i] = rsqrtf(deg_dinv[i] + 1.0f);
}

__global__ void k_norm(const int* __restrict__ src, const int* __restrict__ dst,
                       const float* __restrict__ dinv, float* __restrict__ norm, int e) {
    int i = blockIdx.x * blockDim.x + threadIdx.x;
    if (i < e) norm[i] = dinv[src[i]] * dinv[dst[i]];
}

// ---------------- bf16 WMMA GEMM:  G[n,j] = sum_k h[n,k] * W[j,k] ----------------
// Block = 8 waves, tile = 16 rows x 128 cols. The 16x256 bf16 A-tile (8 KB,
// contiguous in hb) is staged into LDS with async global->LDS b128 copies
// (ASYNCcnt), then all 8 waves consume it via ds_load; each wave does
// 8 x v_wmma_f32_16x16x32_bf16 for its own 16x16 output tile.
__global__ __launch_bounds__(256)
void k_gemm_bf16(const __bf16* __restrict__ h,   // [NN, FD] row-major, bf16
                 const __bf16* __restrict__ W,   // [FD, FD] row-major (W[j,k]), bf16
                 float* __restrict__ G)          // [NN, FD] f32
{
    __shared__ __bf16 sA[16 * APITCH];           // 16 rows, 528B pitch (8448 B)

    const int t    = threadIdx.x;                // 0..255
    const int wave = t >> 5;                     // 0..7
    const int lane = t & 31;
    const int half = lane >> 4;                  // 0 | 1
    const int l    = lane & 15;
    const int row0 = blockIdx.y << 4;            // node tile (16 rows)
    const int col0 = (blockIdx.x << 7) + (wave << 4); // this wave's 16 cols

    // ---- async stage: 8 KB contiguous h-tile -> padded LDS rows ----
    {
        const int srow = t >> 5;                 // 0..7   (row within tile)
        const int schk = t & 31;                 // 0..31  (16B chunk in 512B row)
        unsigned long long gbase =
            (unsigned long long)(size_t)(h + (size_t)row0 * FD);
        unsigned lbase = (unsigned)(size_t)sA;

        unsigned long long g0 = gbase + (unsigned)(srow * (FD * 2) + schk * 16);
        unsigned           l0 = lbase + (unsigned)(srow * (APITCH * 2) + schk * 16);
        unsigned long long g1 = g0 + 8u * (FD * 2);        // rows 8..15
        unsigned           l1 = l0 + 8u * (APITCH * 2);

        asm volatile("global_load_async_to_lds_b128 %0, %1, off"
                     :: "v"(l0), "v"(g0) : "memory");
        asm volatile("global_load_async_to_lds_b128 %0, %1, off"
                     :: "v"(l1), "v"(g1) : "memory");
        asm volatile("s_wait_asynccnt 0x0" ::: "memory");
        __syncthreads();
    }

    const __bf16* wrow = W + (size_t)(col0 + l) * FD;
    const __bf16* arow = sA + l * APITCH;        // A row M=l (LDS)

    v8f acc = {};
#pragma unroll
    for (int k = 0; k < FD; k += 32) {
        // A 16x32 bf16: lanes0-15 elems0-15=K0..15
        //               lanes16-31 elems0-15=K16..31 (matches B)
        const __bf16* pa0 = arow + k + 16 * half;
        const __bf16* pa1 = arow + k + 16 * half + 8;
        // B 32x16 bf16: lanes0-15 hold K0..15 of column N=l, lanes16-31 K16..31
        const __bf16* pb  = wrow + k + 16 * half;
        v16bf a, b;
#pragma unroll
        for (int i = 0; i < 8; ++i) {
            a[i]     = pa0[i];
            a[8 + i] = pa1[i];
        }
#pragma unroll
        for (int i = 0; i < 16; ++i) b[i] = pb[i];

        acc = __builtin_amdgcn_wmma_f32_16x16x32_bf16(
                  /*neg_a=*/false, a, /*neg_b=*/false, b,
                  /*c_mod=*/(short)0, acc, /*reuse_a=*/false, /*reuse_b=*/false);
    }

    // C/D layout: VGPR r -> lanes0-15: M=r, lanes16-31: M=r+8 ; N = l
#pragma unroll
    for (int r = 0; r < 8; ++r) {
        G[(size_t)(row0 + r + 8 * half) * FD + col0 + l] = acc[r];
    }
}

// ---------------- edge scatter:  agg[dst] += norm[e] * G[src] ----------------
// 64 lanes per edge, float4 per lane (256 features).
__global__ void k_edge_agg(const float* __restrict__ G,
                           const int* __restrict__ src, const int* __restrict__ dst,
                           const float* __restrict__ norm, float* __restrict__ agg, int e)
{
    int t = blockIdx.x * blockDim.x + threadIdx.x;
    int ed = t >> 6;
    if (ed >= e) return;
    int c  = (t & 63) << 2;
    int s  = src[ed];
    int d  = dst[ed];
    float w = norm[ed];
    const float4 v = *(const float4*)(G + (size_t)s * FD + c);
    float* ap = agg + (size_t)d * FD + c;
    unsafeAtomicAdd(ap + 0, v.x * w);
    unsafeAtomicAdd(ap + 1, v.y * w);
    unsafeAtomicAdd(ap + 2, v.z * w);
    unsafeAtomicAdd(ap + 3, v.w * w);
}

// ---------------- self-loop + bias + ReLU ----------------
// out = relu(agg + dinv[n]^2 * G[n] + b); write bf16 for next layer or f32 final.
__global__ void k_finish(const float* __restrict__ agg, const float* __restrict__ G,
                         const float* __restrict__ dinv, const float* __restrict__ bias,
                         float* __restrict__ outf, __bf16* __restrict__ outb, int nh)
{
    int t = blockIdx.x * blockDim.x + threadIdx.x;
    if (t >= nh) return;
    int n = t >> 8;          // FD == 256
    int j = t & (FD - 1);
    float di = dinv[n];
    float v  = agg[t] + di * di * G[t] + bias[j];
    v = fmaxf(v, 0.0f);
    if (outf) outf[t] = v;
    else      outb[t] = (__bf16)v;
}

// ---------------- host ----------------

extern "C" void kernel_launch(void* const* d_in, const int* in_sizes, int n_in,
                              void* d_out, int out_size, void* d_ws, size_t ws_size,
                              hipStream_t stream) {
    const float* x   = (const float*)d_in[0];
    const int*   ei  = (const int*)d_in[1];
    const float* W1  = (const float*)d_in[2];
    const float* b1  = (const float*)d_in[3];
    const float* W2  = (const float*)d_in[4];
    const float* b2  = (const float*)d_in[5];
    const float* W3  = (const float*)d_in[6];
    const float* b3  = (const float*)d_in[7];
    float* out = (float*)d_out;

    const int* src = ei;
    const int* dst = ei + NE;

    char* ws = (char*)d_ws;
    // layout (bytes):
    float*  dinv = (float*) (ws + 0);                 //  40 KB  (deg -> dinv in place)
    float*  norm = (float*) (ws + (1u << 16));        //  1.28 MB
    __bf16* hb   = (__bf16*)(ws + (2u << 20));        //  5.12 MB (bf16 GEMM input)
    __bf16* Wb1  = (__bf16*)(ws + (8u << 20));        //  128 KB each
    __bf16* Wb2  = Wb1 + FD * FD;
    __bf16* Wb3  = Wb2 + FD * FD;
    float*  G    = (float*) (ws + (9u << 20));        // 10.24 MB (GEMM output)
    float*  agg  = (float*) (ws + (20u << 20));       // 10.24 MB (scatter target)

    const int NH = NN * FD;
    dim3 blk256(256);
    dim3 gemm_grid(FD / 128, NN / 16);                // (2, 625) block tiles
    dim3 gemm_blk(256);                               // 8 waves per block
    int gE   = (NE + 255) / 256;
    int gN   = (NN + 255) / 256;
    int gNH  = (NH + 255) / 256;
    int gW   = (FD * FD + 255) / 256;
    int gEf  = (NE * 64 + 255) / 256;                 // 64 lanes per edge

    // ---- normalization precompute ----
    k_zero<<<gN, blk256, 0, stream>>>(dinv, NN);
    k_degree<<<gE, blk256, 0, stream>>>(dst, dinv, NE);
    k_dinv<<<gN, blk256, 0, stream>>>(dinv, NN);
    k_norm<<<gE, blk256, 0, stream>>>(src, dst, dinv, norm, NE);

    // ---- bf16 conversions (weights once, x once) ----
    k_cvt_bf16<<<gW, blk256, 0, stream>>>(W1, Wb1, FD * FD);
    k_cvt_bf16<<<gW, blk256, 0, stream>>>(W2, Wb2, FD * FD);
    k_cvt_bf16<<<gW, blk256, 0, stream>>>(W3, Wb3, FD * FD);
    k_cvt_bf16<<<gNH, blk256, 0, stream>>>(x, hb, NH);

    // ---- layer 1 ----
    k_gemm_bf16<<<gemm_grid, gemm_blk, 0, stream>>>(hb, Wb1, G);
    k_zero<<<gNH, blk256, 0, stream>>>(agg, NH);
    k_edge_agg<<<gEf, blk256, 0, stream>>>(G, src, dst, norm, agg, NE);
    k_finish<<<gNH, blk256, 0, stream>>>(agg, G, dinv, b1, nullptr, hb, NH);

    // ---- layer 2 ----
    k_gemm_bf16<<<gemm_grid, gemm_blk, 0, stream>>>(hb, Wb2, G);
    k_zero<<<gNH, blk256, 0, stream>>>(agg, NH);
    k_edge_agg<<<gEf, blk256, 0, stream>>>(G, src, dst, norm, agg, NE);
    k_finish<<<gNH, blk256, 0, stream>>>(agg, G, dinv, b2, nullptr, hb, NH);

    // ---- layer 3 (float output) ----
    k_gemm_bf16<<<gemm_grid, gemm_blk, 0, stream>>>(hb, Wb3, G);
    k_zero<<<gNH, blk256, 0, stream>>>(agg, NH);
    k_edge_agg<<<gEf, blk256, 0, stream>>>(G, src, dst, norm, agg, NE);
    k_finish<<<gNH, blk256, 0, stream>>>(agg, G, dinv, b3, out, nullptr, NH);
}
